// MutiheadSelfAttention_67104569032838
// MI455X (gfx1250) — compile-verified, matrix operand order checked by reading, not run on hardware
//
#include <hip/hip_runtime.h>

// ---------------------------------------------------------------------------
// MHSA for MI455X (gfx1250): bf16 WMMA + async-to-LDS staging.
//   B=16, N=1024, D=1024, H=8, HD=128, SCALE=1/sqrt(128), mask fill = -1e30
// ---------------------------------------------------------------------------

typedef __attribute__((ext_vector_type(16))) __bf16 v16bf;
typedef __attribute__((ext_vector_type(8)))  __bf16 v8bf;
typedef __attribute__((ext_vector_type(8)))  float  v8f;

#define BATCH  16
#define SEQ    1024
#define DIM    1024
#define NHEAD  8
#define HDIM   128
#define ATTN_SCALE 0.08838834764831845f   // 1/sqrt(128)
#define NEGINF (-1.0e30f)

// D = A(16x32 bf16) * B(32x16 bf16) + C(16x16 f32)
__device__ __forceinline__ v8f wmma_bf16(v16bf a, v16bf b, v8f c) {
  return __builtin_amdgcn_wmma_f32_16x16x32_bf16(
      /*neg_a=*/false, a, /*neg_b=*/false, b,
      /*c_mod=*/(short)0, c, /*reuse_a=*/false, /*reuse_b=*/false);
}

// Load a 16x32 bf16 fragment (A layout; also B layout when the source is
// K-major, i.e. Bt[n][k]).  Per CDNA5 ISA 7.12.2:
//   lanes 0-15 : row = lane,    k = {0..7} U {16..23}
//   lanes16-31 : row = lane-16, k = {8..15} U {24..31}
// Each half is 8 contiguous bf16 = one 16-byte load (b128).
__device__ __forceinline__ v16bf load_frag(const __bf16* src, int ld) {
  int lane = threadIdx.x & 31;
  int row  = lane & 15;
  int kb   = (lane & 16) ? 8 : 0;
  const __bf16* p = src + (long)row * ld + kb;
  v8bf lo = *(const v8bf*)(p);
  v8bf hi = *(const v8bf*)(p + 16);
  v16bf r;
#pragma unroll
  for (int i = 0; i < 8; ++i) { r[i] = lo[i]; r[8 + i] = hi[i]; }
  return r;
}

// Async copy of 16 bytes global -> LDS (gfx1250, tracked by ASYNCcnt).
// VDST VGPR carries the per-lane LDS byte address (low 32 bits of the
// generic pointer are the LDS offset per the flat-aperture rules).
__device__ __forceinline__ void async_copy16(const __bf16* g, __bf16* l) {
  unsigned lds = (unsigned)(unsigned long long)l;
  asm volatile("global_load_async_to_lds_b128 %0, %1, off"
               :: "v"(lds), "v"(g) : "memory");
}
__device__ __forceinline__ void wait_async0() {
  asm volatile("s_wait_asynccnt 0x0" ::: "memory");
}

// ---------------------------------------------------------------------------
// Elementwise conversion kernels
// ---------------------------------------------------------------------------
__global__ void f32_to_bf16_kernel(const float* __restrict__ in,
                                   __bf16* __restrict__ out, long n) {
  long i  = (long)blockIdx.x * blockDim.x + threadIdx.x;
  long st = (long)gridDim.x * blockDim.x;
  for (; i < n; i += st) out[i] = (__bf16)in[i];
}

// W[b][r][c] (f32) -> Wt[b][c][r] (bf16): K-major weights for B fragments.
__global__ void transpose_f32_to_bf16_kernel(const float* __restrict__ in,
                                             __bf16* __restrict__ out,
                                             int batch, int R, int C) {
  long n  = (long)batch * R * C;
  long i  = (long)blockIdx.x * blockDim.x + threadIdx.x;
  long st = (long)gridDim.x * blockDim.x;
  for (; i < n; i += st) {
    long bi  = i / ((long)R * C);
    long rem = i - bi * (long)R * C;
    int r = (int)(rem / C);
    int c = (int)(rem % C);
    out[bi * (long)R * C + (long)c * R + r] = (__bf16)in[i];
  }
}

// ---------------------------------------------------------------------------
// Generic batched GEMM:  out[batch] = A(MxK) * Bt(NnxK)^T + bias
// Block = 4 waves covering a 64x64 output tile.  The 64x32 B tile is shared
// by all 4 waves, so it is staged once per block into LDS with async copies
// (double-buffered: stage k+32 while WMMAs consume k).
// M, Nn must be multiples of 64 here.
// ---------------------------------------------------------------------------
__global__ void __launch_bounds__(128)
gemm_bf16_wmma(const __bf16* __restrict__ A, long long strideA,
               const __bf16* __restrict__ Bt, long long strideB,
               const float*  __restrict__ bias, long long strideBias,
               __bf16* __restrict__ outBf, float* __restrict__ outF32,
               long long strideOut, int ldOut, int transposeOut,
               int M, int Nn, int K, int Hdiv) {
  __shared__ __bf16 sB[2][64 * 32];     // two 4KB B-tile buffers

  int wave = threadIdx.x >> 5;
  int lane = threadIdx.x & 31;
  int tm   = blockIdx.x * 4 + wave;
  int batch = blockIdx.z;
  const __bf16* Ab = A  + (long)(batch / Hdiv) * strideA + (long)tm * 16 * K;
  const __bf16* Bb = Bt + (long)(batch % Hdiv) * strideB;
  const float* biasb = bias ? bias + (long)(batch % Hdiv) * strideBias : nullptr;
  int tn0 = blockIdx.y * 64;

  // Stage one 64(cols) x 32(k) tile of Bt into sB[buf]: 256 x 16B chunks,
  // 2 per thread, all issued asynchronously.
  auto stageB = [&](int buf, int kc) {
#pragma unroll
    for (int i = 0; i < 2; ++i) {
      int idx  = threadIdx.x + i * 128;     // 0..255
      int row  = idx >> 2;                  // 0..63  (column of the output)
      int part = (idx & 3) * 8;             // element offset within k-run
      async_copy16(Bb + (long)(tn0 + row) * K + kc + part,
                   &sB[buf][row * 32 + part]);
    }
  };

  stageB(0, 0);
  wait_async0();
  __syncthreads();

  v8f c0 = {}, c1 = {}, c2 = {}, c3 = {};
  int buf = 0;
  for (int kc = 0; kc < K; kc += 32) {
    if (kc + 32 < K) stageB(buf ^ 1, kc + 32);

    v16bf a  = load_frag(Ab + kc, K);
    v16bf b0 = load_frag(&sB[buf][ 0 * 32], 32);
    v16bf b1 = load_frag(&sB[buf][16 * 32], 32);
    v16bf b2 = load_frag(&sB[buf][32 * 32], 32);
    v16bf b3 = load_frag(&sB[buf][48 * 32], 32);
    c0 = wmma_bf16(a, b0, c0);
    c1 = wmma_bf16(a, b1, c1);
    c2 = wmma_bf16(a, b2, c2);
    c3 = wmma_bf16(a, b3, c3);

    wait_async0();          // next tile landed in LDS
    __syncthreads();        // visible to all waves; prev reads complete
    buf ^= 1;
  }

  int half8 = (lane >> 4) & 1;
  int col   = lane & 15;
  v8f cc[4] = {c0, c1, c2, c3};
#pragma unroll
  for (int f = 0; f < 4; ++f) {
    int gc = tn0 + f * 16 + col;
    float bv = biasb ? biasb[gc] : 0.0f;
#pragma unroll
    for (int v = 0; v < 8; ++v) {
      int gr = tm * 16 + v + 8 * half8;
      float val = cc[f][v] + bv;
      long off = (long)batch * strideOut +
                 (transposeOut ? ((long)gc * ldOut + gr)
                               : ((long)gr * ldOut + gc));
      if (outBf) outBf[off] = (__bf16)val;
      else       outF32[off] = val;
    }
  }
}

// ---------------------------------------------------------------------------
// Flash attention: waves are fully independent (16 q-rows each).
// Q,K: bf16 [B*H][SEQ][HDIM].  Vt: bf16 [B*H][HDIM][SEQ] (pre-transposed by
// the V projection GEMM, so P*V B-fragments are contiguous 16B runs).
// O: bf16 [B][SEQ][DIM] with heads concatenated.
// ---------------------------------------------------------------------------
__global__ void __launch_bounds__(256)
flash_attn_wmma(const __bf16* __restrict__ Q, const __bf16* __restrict__ K,
                const __bf16* __restrict__ Vt, const int* __restrict__ xmask,
                __bf16* __restrict__ O) {
  __shared__ __bf16 sP[8 * 16 * 32];    // per-wave P tile (16 rows x 32 keys)

  int wave  = threadIdx.x >> 5;
  int lane  = threadIdx.x & 31;
  int bh    = blockIdx.x >> 3;          // SEQ/128 = 8 query blocks per (b,h)
  int qblk  = blockIdx.x & 7;
  int b     = bh >> 3;                  // H == 8
  int h     = bh & 7;
  int qi0   = qblk * 128 + wave * 16;

  const __bf16* Qb  = Q  + (long)bh * SEQ * HDIM + (long)qi0 * HDIM;
  const __bf16* Kb  = K  + (long)bh * SEQ * HDIM;
  const __bf16* Vtb = Vt + (long)bh * HDIM * SEQ;
  const int* mrow   = xmask + (long)b * SEQ;

  int half8 = (lane >> 4) & 1;
  int col   = lane & 15;

  // Q fragments (16 rows x HD=128 -> 4 k-chunks), kept in registers.
  v16bf qa[4];
#pragma unroll
  for (int c = 0; c < 4; ++c) qa[c] = load_frag(Qb + c * 32, HDIM);

  int mq[8];
#pragma unroll
  for (int v = 0; v < 8; ++v) mq[v] = mrow[qi0 + v + 8 * half8];

  v8f o[8] = {};                        // 16 x 128 f32 accumulator
  float mr[8], lr[8];
#pragma unroll
  for (int v = 0; v < 8; ++v) { mr[v] = -3.0e38f; lr[v] = 0.0f; }

  __bf16* pw = sP + wave * 512;

  for (int kb = 0; kb < SEQ / 32; ++kb) {
    // Prefetch next K block (one row per lane) -> global_prefetch_b8.
    if (kb + 1 < SEQ / 32)
      __builtin_prefetch(Kb + (long)(kb * 32 + 32 + lane) * HDIM, 0, 3);

    // S = Q * K^T for two 16-key tiles; K rows are already K-major for B.
    v8f s0 = {}, s1 = {};
#pragma unroll
    for (int c = 0; c < 4; ++c) {
      v16bf k0 = load_frag(Kb + (long)(kb * 32)      * HDIM + c * 32, HDIM);
      v16bf k1 = load_frag(Kb + (long)(kb * 32 + 16) * HDIM + c * 32, HDIM);
      s0 = wmma_bf16(qa[c], k0, s0);
      s1 = wmma_bf16(qa[c], k1, s1);
    }

    int mk0 = mrow[kb * 32 + col];
    int mk1 = mrow[kb * 32 + 16 + col];

    float corr[8];
#pragma unroll
    for (int v = 0; v < 8; ++v) {
      bool rowok = (mq[v] != 0);
      float a0 = (rowok && mk0) ? s0[v] * ATTN_SCALE : NEGINF;
      float a1 = (rowok && mk1) ? s1[v] * ATTN_SCALE : NEGINF;
      // row max across the 16 lanes holding this row
      float rmax = fmaxf(a0, a1);
#pragma unroll
      for (int d = 1; d < 16; d <<= 1) rmax = fmaxf(rmax, __shfl_xor(rmax, d, 32));
      float nm = fmaxf(mr[v], rmax);
      float cf = __expf(mr[v] - nm);
      float p0 = __expf(a0 - nm);
      float p1 = __expf(a1 - nm);
      float rs = p0 + p1;
#pragma unroll
      for (int d = 1; d < 16; d <<= 1) rs += __shfl_xor(rs, d, 32);
      lr[v] = lr[v] * cf + rs;
      mr[v] = nm;
      corr[v] = cf;
      int rv = v + 8 * half8;
      pw[rv * 32 + col]      = (__bf16)p0;   // C layout -> row-major P in LDS
      pw[rv * 32 + 16 + col] = (__bf16)p1;
    }

    // Rescale running output by the max-correction factor.
#pragma unroll
    for (int f = 0; f < 8; ++f)
#pragma unroll
      for (int v = 0; v < 8; ++v) o[f][v] *= corr[v];

    // O += P(16x32) * V(32x128): A frag from LDS, 8 B frags from global Vt.
    v16bf pa = load_frag(pw, 32);
#pragma unroll
    for (int f = 0; f < 8; ++f) {
      v16bf vb = load_frag(Vtb + (long)(f * 16) * SEQ + kb * 32, SEQ);
      o[f] = wmma_bf16(pa, vb, o[f]);
    }
  }

  // Normalize and store with heads concatenated: O[b][row][h*128 + col].
  float inv[8];
#pragma unroll
  for (int v = 0; v < 8; ++v) inv[v] = 1.0f / lr[v];
#pragma unroll
  for (int f = 0; f < 8; ++f) {
#pragma unroll
    for (int v = 0; v < 8; ++v) {
      int rv = v + 8 * half8;
      long off = ((long)b * SEQ + qi0 + rv) * DIM + h * HDIM + f * 16 + col;
      O[off] = (__bf16)(o[f][v] * inv[v]);
    }
  }
}

// ---------------------------------------------------------------------------
// Host-side launcher
// ---------------------------------------------------------------------------
extern "C" void kernel_launch(void* const* d_in, const int* in_sizes, int n_in,
                              void* d_out, int out_size, void* d_ws, size_t ws_size,
                              hipStream_t stream) {
  const float* x   = (const float*)d_in[0];
  const int*   xm  = (const int*)d_in[1];
  const float* Wq  = (const float*)d_in[2];
  const float* bq  = (const float*)d_in[3];
  const float* Wk  = (const float*)d_in[4];
  const float* bk  = (const float*)d_in[5];
  const float* Wv  = (const float*)d_in[6];
  const float* bv  = (const float*)d_in[7];
  const float* Wfc = (const float*)d_in[8];
  const float* bfc = (const float*)d_in[9];
  float* out = (float*)d_out;

  char* wsp = (char*)d_ws;
  auto alloc = [&](size_t bytes) -> char* {
    char* p = wsp;
    wsp += (bytes + 255) & ~(size_t)255;
    return p;
  };
  __bf16* xbf  = (__bf16*)alloc((size_t)BATCH * SEQ * DIM * 2);
  __bf16* wqT  = (__bf16*)alloc((size_t)NHEAD * HDIM * DIM * 2);
  __bf16* wkT  = (__bf16*)alloc((size_t)NHEAD * HDIM * DIM * 2);
  __bf16* wvT  = (__bf16*)alloc((size_t)NHEAD * HDIM * DIM * 2);
  __bf16* wfcT = (__bf16*)alloc((size_t)DIM * DIM * 2);
  __bf16* qbf  = (__bf16*)alloc((size_t)BATCH * NHEAD * SEQ * HDIM * 2);
  __bf16* kbf  = (__bf16*)alloc((size_t)BATCH * NHEAD * SEQ * HDIM * 2);
  __bf16* vtbf = (__bf16*)alloc((size_t)BATCH * NHEAD * HDIM * SEQ * 2);
  __bf16* obf  = (__bf16*)alloc((size_t)BATCH * SEQ * DIM * 2);

  // Stage 1: precision conversion + weight transposes (K-major B operands).
  f32_to_bf16_kernel<<<4096, 256, 0, stream>>>(x, xbf, (long)BATCH * SEQ * DIM);
  transpose_f32_to_bf16_kernel<<<2048, 256, 0, stream>>>(Wq, wqT, NHEAD, DIM, HDIM);
  transpose_f32_to_bf16_kernel<<<2048, 256, 0, stream>>>(Wk, wkT, NHEAD, DIM, HDIM);
  transpose_f32_to_bf16_kernel<<<2048, 256, 0, stream>>>(Wv, wvT, NHEAD, DIM, HDIM);
  transpose_f32_to_bf16_kernel<<<2048, 256, 0, stream>>>(Wfc, wfcT, 1, DIM, DIM);

  // Stage 2: QKV projections, batched over B*H (Hdiv=NHEAD splits A vs W/bias).
  // V is written TRANSPOSED (key-major) for the attention P*V WMMAs.
  dim3 gq(SEQ / 64, HDIM / 64, BATCH * NHEAD);
  gemm_bf16_wmma<<<gq, 128, 0, stream>>>(
      xbf, (long long)SEQ * DIM, wqT, (long long)HDIM * DIM, bq, (long long)HDIM,
      qbf, nullptr, (long long)SEQ * HDIM, HDIM, 0, SEQ, HDIM, DIM, NHEAD);
  gemm_bf16_wmma<<<gq, 128, 0, stream>>>(
      xbf, (long long)SEQ * DIM, wkT, (long long)HDIM * DIM, bk, (long long)HDIM,
      kbf, nullptr, (long long)SEQ * HDIM, HDIM, 0, SEQ, HDIM, DIM, NHEAD);
  gemm_bf16_wmma<<<gq, 128, 0, stream>>>(
      xbf, (long long)SEQ * DIM, wvT, (long long)HDIM * DIM, bv, (long long)HDIM,
      vtbf, nullptr, (long long)SEQ * HDIM, SEQ, 1, SEQ, HDIM, DIM, NHEAD);

  // Stage 3: masked flash attention, heads concatenated into obf.
  flash_attn_wmma<<<BATCH * NHEAD * (SEQ / 128), 256, 0, stream>>>(
      qbf, kbf, vtbf, xm, obf);

  // Stage 4: final FC straight into f32 d_out.
  dim3 gf((BATCH * SEQ) / 64, DIM / 64, 1);
  gemm_bf16_wmma<<<gf, 128, 0, stream>>>(
      obf, 0, wfcT, 0, bfc, 0,
      nullptr, out, 0, DIM, 0, BATCH * SEQ, DIM, DIM, 1);
}
